// TemporalEncoder_56066503082143
// MI455X (gfx1250) — matrix operand order repaired and checked by reading, not compiled, not run on hardware
//
#include <hip/hip_runtime.h>

// ---------------------------------------------------------------------------
// TemporalEncoder for MI455X (gfx1250, wave32, WMMA).
//   conv1(1->16) scalar -> conv2(16->32) as WMMA GEMM over flat (seq,t) rows
//   (im2col via per-sequence zero apron, LDS-transposed coalesced stores)
//   -> x-proj + GRU(H=64) recurrence fully on v_wmma_f32_16x16x32_f16.
// Weights live in VGPRs across the recurrence; h lives in LDS; gate math uses
// native transcendentals; ReLU is a single v_med3_f32.
// ---------------------------------------------------------------------------

typedef _Float16 half_t;
typedef __attribute__((ext_vector_type(16))) _Float16 v16h;
typedef __attribute__((ext_vector_type(8)))  _Float16 v8h;
typedef __attribute__((ext_vector_type(8)))  float    v8f;

#define WMMA_F16(a, b, c) \
  __builtin_amdgcn_wmma_f32_16x16x32_f16(false, (a), false, (b), (short)0, (c), false, false)

constexpr int BB = 16, NN = 512, TT = 100, HH = 64;
constexpr int BN = BB * NN;        // 8192 sequences
constexpr int TA = TT + 2;         // conv1 rows incl. 2-row causal zero apron
constexpr int C1 = 16, C2 = 32;
constexpr int G3 = 3 * HH;         // 192 gate columns (r | z | n)
constexpr int C2P = C2 + 2;        // padded LDS row stride (bank spreading)

// single-instruction ReLU: v_med3_f32(x, 0, +inf)
__device__ __forceinline__ float relu1(float x) {
  return __builtin_amdgcn_fmed3f(x, 0.0f, __builtin_huge_valf());
}

// --- fast transcendentals: single TRANS ops, no IEEE divide sequences -------
__device__ __forceinline__ float fast_tanh(float x) {
#if __has_builtin(__builtin_amdgcn_tanhf)
  return __builtin_amdgcn_tanhf(x);
#elif __has_builtin(__builtin_amdgcn_tanh_f32)
  return __builtin_amdgcn_tanh_f32(x);
#else
  return 2.0f * __builtin_amdgcn_rcpf(1.0f + __expf(-2.0f * x)) - 1.0f;
#endif
}
__device__ __forceinline__ float fast_sig(float x) {
#if __has_builtin(__builtin_amdgcn_tanhf) || __has_builtin(__builtin_amdgcn_tanh_f32)
  return fmaf(0.5f, fast_tanh(0.5f * x), 0.5f);
#else
  return __builtin_amdgcn_rcpf(1.0f + __expf(-x));
#endif
}

// ---------------------------------------------------------------------------
// Kernel 0: weight prep (fp32 -> fp16, WMMA-B layouts).
// ---------------------------------------------------------------------------
__global__ void prep_weights_kernel(const float* __restrict__ wih,
                                    const float* __restrict__ whh,
                                    const float* __restrict__ w2,
                                    half_t* __restrict__ wih_h,
                                    half_t* __restrict__ whh_h,
                                    half_t* __restrict__ w2h) {
  int i = blockIdx.x * blockDim.x + threadIdx.x;
  int isw = (i & ~24) | ((i & 8) << 1) | ((i & 16) >> 1);
  if (i < G3 * C2) wih_h[i] = (half_t)wih[isw];
  if (i < G3 * HH) whh_h[i] = (half_t)whh[isw];
  if (i < C2 * 64) {  // conv2 B operand [o][k], k = tap*16+ch, K padded to 64
    int o = i >> 6, k = isw & 63, tap = k >> 4, ch = k & 15;
    w2h[i] = (tap < 3) ? (half_t)w2[(o * C1 + ch) * 3 + tap] : (half_t)0.0f;
  }
}

// ---------------------------------------------------------------------------
// Kernel 1: conv1 (1->16, causal, relu), fp16 out with 2-row zero apron:
//   c1h[seq][r][16], r in [0,102): r<2 -> zeros, else conv1 at time r-2.
// ---------------------------------------------------------------------------
__global__ __launch_bounds__(256)
void conv1_kernel(const float* __restrict__ x,
                  const float* __restrict__ w1, const float* __restrict__ b1,
                  half_t* __restrict__ c1h) {
  int idx = blockIdx.x * 256 + threadIdx.x;      // flat (seq, apron row)
  if (idx >= BN * TA) return;
  int r = idx % TA;
  int seq = idx / TA;

  half_t o16[C1];
  if (r < 2) {
#pragma unroll
    for (int o = 0; o < C1; ++o) o16[o] = (half_t)0.0f;
  } else {
    int t = r - 2;
    const float* xr = x + (size_t)seq * TT;
    float x0 = (t >= 2) ? xr[t - 2] : 0.0f;
    float x1 = (t >= 1) ? xr[t - 1] : 0.0f;
    float x2 = xr[t];
#pragma unroll
    for (int o = 0; o < C1; ++o) {
      float a = b1[o] + w1[o * 3] * x0 + w1[o * 3 + 1] * x1 + w1[o * 3 + 2] * x2;
      o16[o] = (half_t)relu1(a);
    }
  }
  half_t* dst = c1h + (size_t)idx * C1;
  *(v8h*)(dst)     = *(const v8h*)(o16);
  *(v8h*)(dst + 8) = *(const v8h*)(o16 + 8);
}

// ---------------------------------------------------------------------------
// Kernel 2: conv2 (16->32, causal, relu) as WMMA GEMM.
//   rows = 16 consecutive FLAT (seq,t) indices (819200 % 16 == 0 -> no tail,
//          no masking, EXEC all-ones throughout)
//   K=64 = im2col [c1(t-2)|c1(t-1)|c1(t)|zeros], two K=32 WMMAs
//   N=32 = two 16-column output tiles
// D fragments are transposed through a per-wave LDS slice so the global write
// is 2 coalesced b128 stores per lane instead of 16 scattered b16 stores.
// ---------------------------------------------------------------------------
__global__ __launch_bounds__(128)
void conv2_wmma_kernel(const half_t* __restrict__ c1h,
                       const float* __restrict__ b2,
                       const half_t* __restrict__ w2h,
                       half_t* __restrict__ seqh) {
  __shared__ half_t xpose[4][16 * C2P];          // per-wave private slices

  const int tid  = threadIdx.x;
  const int wv   = tid >> 5;
  const int lane = tid & 31;
  const int m    = lane & 15;
  const int hi   = lane >> 4;
  const int row0 = (blockIdx.x * 4 + wv) * 16;   // flat row tile base

  // B tiles: w2h is [o][k], k bits 3,4 swapped; lane = col o = nt*16+m, K c*32+8hi+{0..7,16..23}.
  v16h B[2][2];
#pragma unroll
  for (int nt = 0; nt < 2; ++nt)
#pragma unroll
    for (int c = 0; c < 2; ++c)
      B[nt][c] = *(const v16h*)(w2h + (size_t)(nt * 16 + m) * 64 + c * 32 + hi * 16);

  // A row for lane m: flat row -> (seq, t); apron row t holds conv1(t-2).
  const int idx = row0 + m;
  const int seq = idx / TT;
  const int t   = idx - seq * TT;
  const half_t* ar = c1h + ((size_t)seq * TA + t) * C1;
  v8h a0 = *(const v8h*)(ar + hi * 8);            // tap t-2 -> K base..+7
  v8h a1 = *(const v8h*)(ar + C1 + hi * 8);       // tap t-1 -> K base+16..+23
  v8h a2 = *(const v8h*)(ar + 2 * C1 + hi * 8);   // tap t   -> K 32+base..
  v16h A0, A1;
#pragma unroll
  for (int i = 0; i < 8; ++i) {
    A0[i] = a0[i]; A0[i + 8] = a1[i];
    A1[i] = a2[i]; A1[i + 8] = (half_t)0.0f;      // K 48..63 zero pad
  }

  const float b0 = b2[m], b1v = b2[16 + m];
  v8f acc0 = {b0, b0, b0, b0, b0, b0, b0, b0};
  v8f acc1 = {b1v, b1v, b1v, b1v, b1v, b1v, b1v, b1v};
  acc0 = WMMA_F16(A0, B[0][0], acc0);
  acc0 = WMMA_F16(A1, B[0][1], acc0);
  acc1 = WMMA_F16(A0, B[1][0], acc1);
  acc1 = WMMA_F16(A1, B[1][1], acc1);

  // D element i = (row r = i+8*hi, cols m and 16+m): relu+cvt into LDS slice.
  half_t* xp = xpose[wv];
#pragma unroll
  for (int i = 0; i < 8; ++i) {
    int r = i + hi * 8;
    xp[r * C2P + m]      = (half_t)relu1(acc0[i]);
    xp[r * C2P + 16 + m] = (half_t)relu1(acc1[i]);
  }
  // LDS ops of a wave are in-order (DScnt): read back row-major, store b128s.
  v8h r0 = *(const v8h*)(xp + m * C2P + hi * 16);
  v8h r1 = *(const v8h*)(xp + m * C2P + hi * 16 + 8);
  half_t* dst = seqh + (size_t)idx * C2 + hi * 16;
  *(v8h*)(dst)     = r0;
  *(v8h*)(dst + 8) = r1;
}

// ---------------------------------------------------------------------------
// Kernel 3: GRU recurrence. 512 blocks x 128 threads (4 wave32).
// Block = 16 sequences; wave w owns h-columns [16w,16w+16): 9 WMMAs/step
// (3 x-proj K=32, 6 h-proj K=64 as 2xK=32; n-gate x/h separate for
// n = tanh(xn + r*hn)). Elementwise update is register-local; h_t moves
// through 2KB LDS with two barriers per step.
// ---------------------------------------------------------------------------
__global__ __launch_bounds__(128)
void gru_kernel(const half_t* __restrict__ seqh,
                const half_t* __restrict__ wih_h,
                const half_t* __restrict__ whh_h,
                const float* __restrict__ b_ih,
                const float* __restrict__ b_hh,
                float* __restrict__ out) {
  __shared__ half_t hbuf[16 * HH];

  const int tid  = threadIdx.x;
  const int w    = tid >> 5;
  const int lane = tid & 31;
  const int m    = lane & 15;
  const int hi   = lane >> 4;
  const int seqBase = blockIdx.x * 16;
  const int jcol = 16 * w + m;

  v16h Bih[3];
  v16h Bhh[3][2];
#pragma unroll
  for (int g = 0; g < 3; ++g) {
    int grow = g * 64 + jcol;
    Bih[g]    = *(const v16h*)(wih_h + (size_t)grow * 32 + hi * 16);
    Bhh[g][0] = *(const v16h*)(whh_h + (size_t)grow * 64 + hi * 16);
    Bhh[g][1] = *(const v16h*)(whh_h + (size_t)grow * 64 + 32 + hi * 16);
  }
  const float br  = b_ih[jcol]      + b_hh[jcol];
  const float bz  = b_ih[64 + jcol] + b_hh[64 + jcol];
  const float bnx = b_ih[128 + jcol];
  const float bnh = b_hh[128 + jcol];

#pragma unroll
  for (int q = 0; q < 8; ++q) hbuf[tid * 8 + q] = (half_t)0.0f;
  float hp[8];
#pragma unroll
  for (int i = 0; i < 8; ++i) hp[i] = 0.0f;
  __syncthreads();

  const half_t* xrow = seqh + (size_t)(seqBase + m) * TT * C2;

  for (int t = 0; t < TT; ++t) {
    v8h a0 = *(const v8h*)(xrow + (size_t)t * C2 + hi * 8);
    v8h a1 = *(const v8h*)(xrow + (size_t)t * C2 + hi * 8 + 16);
    v16h xA;
#pragma unroll
    for (int i = 0; i < 8; ++i) { xA[i] = a0[i]; xA[i + 8] = a1[i]; }

    if (t + 1 < TT)
      __builtin_prefetch(xrow + (size_t)(t + 1) * C2, 0, 3);

    v8h h0 = *(const v8h*)(hbuf + m * 64 + hi * 8);
    v8h h1 = *(const v8h*)(hbuf + m * 64 + hi * 8 + 16);
    v8h h2 = *(const v8h*)(hbuf + m * 64 + 32 + hi * 8);
    v8h h3 = *(const v8h*)(hbuf + m * 64 + 32 + hi * 8 + 16);
    v16h hA0, hA1;
#pragma unroll
    for (int i = 0; i < 8; ++i) {
      hA0[i] = h0[i]; hA0[i + 8] = h1[i];
      hA1[i] = h2[i]; hA1[i + 8] = h3[i];
    }

    v8f ar  = {br,  br,  br,  br,  br,  br,  br,  br };
    v8f az  = {bz,  bz,  bz,  bz,  bz,  bz,  bz,  bz };
    v8f anx = {bnx, bnx, bnx, bnx, bnx, bnx, bnx, bnx};
    v8f anh = {bnh, bnh, bnh, bnh, bnh, bnh, bnh, bnh};

    ar  = WMMA_F16(xA,  Bih[0],    ar);
    az  = WMMA_F16(xA,  Bih[1],    az);
    anx = WMMA_F16(xA,  Bih[2],    anx);
    ar  = WMMA_F16(hA0, Bhh[0][0], ar);
    ar  = WMMA_F16(hA1, Bhh[0][1], ar);
    az  = WMMA_F16(hA0, Bhh[1][0], az);
    az  = WMMA_F16(hA1, Bhh[1][1], az);
    anh = WMMA_F16(hA0, Bhh[2][0], anh);
    anh = WMMA_F16(hA1, Bhh[2][1], anh);

#pragma unroll
    for (int i = 0; i < 8; ++i) {
      float r  = fast_sig(ar[i]);
      float z  = fast_sig(az[i]);
      float nn = fast_tanh(fmaf(r, anh[i], anx[i]));
      hp[i] = fmaf(z, hp[i] - nn, nn);     // (1-z)*nn + z*hp
    }

    __syncthreads();
#pragma unroll
    for (int i = 0; i < 8; ++i) {
      int s = i + hi * 8;
      hbuf[s * 64 + jcol] = (half_t)hp[i];
    }
    __syncthreads();
  }

#pragma unroll
  for (int i = 0; i < 8; ++i) {
    int s = i + hi * 8;
    out[(size_t)(seqBase + s) * HH + jcol] = hp[i];
  }
}

// ---------------------------------------------------------------------------
extern "C" void kernel_launch(void* const* d_in, const int* in_sizes, int n_in,
                              void* d_out, int out_size, void* d_ws, size_t ws_size,
                              hipStream_t stream) {
  const float* x   = (const float*)d_in[0];
  const float* w1  = (const float*)d_in[1];
  const float* b1  = (const float*)d_in[2];
  const float* w2  = (const float*)d_in[3];
  const float* b2  = (const float*)d_in[4];
  const float* Wih = (const float*)d_in[5];
  const float* Whh = (const float*)d_in[6];
  const float* bih = (const float*)d_in[7];
  const float* bhh = (const float*)d_in[8];
  float* out = (float*)d_out;

  // workspace layout (fp16): seqh 52.4MB | c1h 26.7MB | weights
  half_t* seqh  = (half_t*)d_ws;
  half_t* c1h   = seqh  + (size_t)BN * TT * C2;
  half_t* wih_h = c1h   + (size_t)BN * TA * C1;
  half_t* whh_h = wih_h + (size_t)G3 * C2;
  half_t* w2h   = whh_h + (size_t)G3 * HH;

  prep_weights_kernel<<<(G3 * HH + 255) / 256, 256, 0, stream>>>(
      Wih, Whh, w2, wih_h, whh_h, w2h);
  conv1_kernel<<<(BN * TA + 255) / 256, 256, 0, stream>>>(x, w1, b1, c1h);
  conv2_wmma_kernel<<<(BN * TT / 16) / 4, 128, 0, stream>>>(c1h, b2, w2h, seqh);
  gru_kernel<<<BN / 16, 128, 0, stream>>>(seqh, wih_h, whh_h, bih, bhh, out);
}
